// OwnSingleHeadTransformer_11003706213051
// MI455X (gfx1250) — matrix operand order repaired and checked by reading, not compiled, not run on hardware
//
#include <hip/hip_runtime.h>
#include <math.h>

// ---------------------------------------------------------------------------
// Types for CDNA5 WMMA (wave32)
// ---------------------------------------------------------------------------
typedef __bf16 bf16_t;
typedef __bf16 v16bf __attribute__((ext_vector_type(16)));
typedef __bf16 v8bf  __attribute__((ext_vector_type(8)));
typedef __bf16 v4bf  __attribute__((ext_vector_type(4)));
typedef float  v8f   __attribute__((ext_vector_type(8)));

static constexpr int cB  = 8;
static constexpr int cS  = 2048;
static constexpr int cD  = 768;
static constexpr int cFF = 3072;

// ---------------------------------------------------------------------------
// f32 -> bf16 conversion (vectorized, 4 elems / thread)
// ---------------------------------------------------------------------------
__global__ void cvt_f32_bf16(const float* __restrict__ in,
                             bf16_t* __restrict__ out, int n4)
{
    int i = blockIdx.x * blockDim.x + threadIdx.x;
    if (i >= n4) return;
    const float4 f = ((const float4*)in)[i];
    v4bf o;
    o[0] = (bf16_t)f.x; o[1] = (bf16_t)f.y;
    o[2] = (bf16_t)f.z; o[3] = (bf16_t)f.w;
    ((v4bf*)out)[i] = o;
}

// ---------------------------------------------------------------------------
// Generic bf16 WMMA GEMM:  C = act( alpha * A[M,K] @ W[N,K]^T + bias[N] )
//   - 256 threads = 8 waves (wave32), block tile 128x128, K-step 32
//   - wave grid 2x4, each wave owns 64x32 (4x2 WMMA 16x16 tiles)
//   - global->LDS staging via async direct-to-LDS copies (ASYNCcnt),
//     double-buffered so tile k+1 streams in while tile k is in the WMMAs
//   - all 6 fragments preloaded before the 8-WMMA batch so the ds_loads
//     pipeline instead of stalling each MAC on s_wait_dscnt
//   - TRANSV: write output transposed per batch of cS rows -> [m/cS][N][cS]
// All dims used here are multiples of the tile sizes -> no bounds checks,
// EXEC stays all-ones around WMMA / async ops as required.
// ---------------------------------------------------------------------------
template <typename OUT_T, bool RELU, bool TRANSV>
__global__ __launch_bounds__(256) void gemm_bf16_wmma(
    const bf16_t* __restrict__ A, const bf16_t* __restrict__ Bm,
    const float* __restrict__ bias, OUT_T* __restrict__ C,
    int M, int N, int K, int lda, int ldb, int ldc,
    long long strideA, long long strideB, long long strideC, float alpha)
{
    constexpr int LP = 40;                      // LDS row pitch (32 + 8 pad)
    __shared__ alignas(16) bf16_t As[2][128 * LP];
    __shared__ alignas(16) bf16_t Bs[2][128 * LP];

    const int tid  = threadIdx.x;
    const int wid  = tid >> 5;
    const int lane = tid & 31;
    const int wm   = wid >> 2;                  // 0..1
    const int wn   = wid & 3;                   // 0..3
    const int half = lane >> 4;                 // lane 0-15 / 16-31
    const int l15  = lane & 15;
    const int tileM = blockIdx.y * 128;
    const int tileN = blockIdx.x * 128;

    const bf16_t* Ab = A  + (long long)blockIdx.z * strideA;
    const bf16_t* Bb = Bm + (long long)blockIdx.z * strideB;

    // staging: each thread copies 2x16B for A and for B per K-step
    const int sr = tid >> 2;                    // 0..63
    const int sk = (tid & 3) * 8;               // 0,8,16,24

    // async global->LDS stage of one 128x32 A tile + 128x32 B tile
    auto stage = [&](int ks, int buf) {
        const int k0 = ks << 5;
        const bf16_t* ga0 = Ab + (size_t)(tileM + sr) * lda + k0 + sk;
        const bf16_t* ga1 = ga0 + (size_t)64 * lda;
        const bf16_t* gb0 = Bb + (size_t)(tileN + sr) * ldb + k0 + sk;
        const bf16_t* gb1 = gb0 + (size_t)64 * ldb;
        const unsigned la0 = (unsigned)(size_t)&As[buf][(sr     ) * LP + sk];
        const unsigned la1 = (unsigned)(size_t)&As[buf][(sr + 64) * LP + sk];
        const unsigned lb0 = (unsigned)(size_t)&Bs[buf][(sr     ) * LP + sk];
        const unsigned lb1 = (unsigned)(size_t)&Bs[buf][(sr + 64) * LP + sk];
        asm volatile(
            "global_load_async_to_lds_b128 %0, %4, off\n\t"
            "global_load_async_to_lds_b128 %1, %5, off\n\t"
            "global_load_async_to_lds_b128 %2, %6, off\n\t"
            "global_load_async_to_lds_b128 %3, %7, off"
            :: "v"(la0), "v"(la1), "v"(lb0), "v"(lb1),
               "v"((unsigned long long)(size_t)ga0),
               "v"((unsigned long long)(size_t)ga1),
               "v"((unsigned long long)(size_t)gb0),
               "v"((unsigned long long)(size_t)gb1)
            : "memory");
    };

    v8f acc[4][2];
    #pragma unroll
    for (int mi = 0; mi < 4; ++mi)
        #pragma unroll
        for (int ni = 0; ni < 2; ++ni)
            #pragma unroll
            for (int e = 0; e < 8; ++e) acc[mi][ni][e] = 0.f;

    const int kSteps = K >> 5;
    stage(0, 0);                                 // prime the pipeline

    for (int ks = 0; ks < kSteps; ++ks) {
        const int buf = ks & 1;
        if (ks + 1 < kSteps) {
            stage(ks + 1, buf ^ 1);              // stream next tile
            // in-order completion: <=4 outstanding means tile ks has landed
            asm volatile("s_wait_asynccnt 0x4" ::: "memory");
        } else {
            asm volatile("s_wait_asynccnt 0x0" ::: "memory");
        }
        __syncthreads();

        // --- preload ALL fragments so the 12 ds_load_b128 pipeline together
        // B fragments: lane half selects a K block of 16, 16 contiguous K
        v16bf bfrag[2];
        #pragma unroll
        for (int ni = 0; ni < 2; ++ni) {
            const bf16_t* p = &Bs[buf][(wn * 32 + ni * 16 + l15) * LP + half * 16];
            const v8bf lo = *(const v8bf*)p;
            const v8bf hi = *(const v8bf*)(p + 8);
            #pragma unroll
            for (int e = 0; e < 8; ++e) { bfrag[ni][e] = lo[e]; bfrag[ni][e + 8] = hi[e]; }
        }
        // A fragments: lane half selects a K block of 16, 16 contiguous K
        v16bf afrag[4];
        #pragma unroll
        for (int mi = 0; mi < 4; ++mi) {
            const bf16_t* p = &As[buf][(wm * 64 + mi * 16 + l15) * LP + half * 16];
            const v8bf lo = *(const v8bf*)p;         // K = kb .. kb+7
            const v8bf hi = *(const v8bf*)(p + 8);   // K = kb+8 .. kb+15
            #pragma unroll
            for (int e = 0; e < 8; ++e) { afrag[mi][e] = lo[e]; afrag[mi][e + 8] = hi[e]; }
        }
        // --- uninterrupted 8-WMMA batch
        #pragma unroll
        for (int mi = 0; mi < 4; ++mi)
            #pragma unroll
            for (int ni = 0; ni < 2; ++ni)
                acc[mi][ni] = __builtin_amdgcn_wmma_f32_16x16x32_bf16(
                    false, afrag[mi], false, bfrag[ni], (short)0, acc[mi][ni],
                    false, false);
        __syncthreads();   // protect buf^1 before it is overwritten next iter
    }

    // --- epilogue: alpha, bias, relu, store (optionally batch-transposed)
    OUT_T* Cb = C + (long long)blockIdx.z * strideC;
    #pragma unroll
    for (int ni = 0; ni < 2; ++ni) {
        const int col = tileN + wn * 32 + ni * 16 + l15;
        const float bv = bias ? bias[col] : 0.f;
        #pragma unroll
        for (int mi = 0; mi < 4; ++mi) {
            const int rbase = tileM + wm * 64 + mi * 16 + half * 8;
            #pragma unroll
            for (int v = 0; v < 8; ++v) {
                float x = acc[mi][ni][v] * alpha + bv;
                if (RELU) x = x > 0.f ? x : 0.f;
                const int row = rbase + v;
                if (TRANSV) {
                    // out[b][col][s], b = row/cS, s = row%cS  (V^T per batch)
                    Cb[((size_t)(row >> 11) * cD + col) * cS + (row & (cS - 1))] =
                        (OUT_T)x;
                } else {
                    Cb[(size_t)row * ldc + col] = (OUT_T)x;
                }
            }
        }
    }
}

// ---------------------------------------------------------------------------
// Row softmax over cS=2048 elements; f32 scores in, bf16 probabilities out.
// 256 threads/row: wave32 shuffle reduce + tiny LDS cross-wave reduce.
// ---------------------------------------------------------------------------
__global__ __launch_bounds__(256) void softmax_rows(const float* __restrict__ Sc,
                                                    bf16_t* __restrict__ P)
{
    const int row = blockIdx.x;
    const float* s = Sc + (size_t)row * cS;
    bf16_t*      p = P  + (size_t)row * cS;
    const int t = threadIdx.x, wid = t >> 5, lane = t & 31;

    float v[8];
    #pragma unroll
    for (int i = 0; i < 8; ++i) v[i] = s[t + (i << 8)];

    float m = v[0];
    #pragma unroll
    for (int i = 1; i < 8; ++i) m = fmaxf(m, v[i]);
    #pragma unroll
    for (int off = 16; off; off >>= 1) m = fmaxf(m, __shfl_xor(m, off, 32));

    __shared__ float red[8];
    if (lane == 0) red[wid] = m;
    __syncthreads();
    #pragma unroll
    for (int i = 0; i < 8; ++i) m = fmaxf(m, red[i]);

    float e[8], sum = 0.f;
    #pragma unroll
    for (int i = 0; i < 8; ++i) { e[i] = __expf(v[i] - m); sum += e[i]; }
    #pragma unroll
    for (int off = 16; off; off >>= 1) sum += __shfl_xor(sum, off, 32);
    __syncthreads();
    if (lane == 0) red[wid] = sum;
    __syncthreads();
    float tot = 0.f;
    #pragma unroll
    for (int i = 0; i < 8; ++i) tot += red[i];
    const float inv = 1.f / tot;
    #pragma unroll
    for (int i = 0; i < 8; ++i) p[t + (i << 8)] = (bf16_t)(e[i] * inv);
}

// ---------------------------------------------------------------------------
// Host-side orchestration
// ---------------------------------------------------------------------------
extern "C" void kernel_launch(void* const* d_in, const int* in_sizes, int n_in,
                              void* d_out, int out_size, void* d_ws, size_t ws_size,
                              hipStream_t stream)
{
    (void)in_sizes; (void)n_in; (void)out_size; (void)ws_size;
    const float* emb = (const float*)d_in[0];
    const float* Wq  = (const float*)d_in[1];
    const float* bq  = (const float*)d_in[2];
    const float* Wk  = (const float*)d_in[3];
    const float* bk  = (const float*)d_in[4];
    const float* Wv  = (const float*)d_in[5];
    const float* bv  = (const float*)d_in[6];
    const float* W1  = (const float*)d_in[7];
    const float* b1  = (const float*)d_in[8];
    const float* W2  = (const float*)d_in[9];
    const float* b2  = (const float*)d_in[10];

    const int M = cB * cS;                       // 16384 token rows

    // workspace carve-up (all offsets 256B aligned; sizes are large pow2-ish)
    char* ws = (char*)d_ws;
    size_t off = 0;
    auto carve = [&](size_t bytes) {
        void* ptr = ws + off;
        off += (bytes + 255) & ~(size_t)255;
        return ptr;
    };
    bf16_t* emb16 = (bf16_t*)carve((size_t)M * cD * 2);      // also reused as attn16
    bf16_t* wq16  = (bf16_t*)carve((size_t)cD * cD * 2);
    bf16_t* wk16  = (bf16_t*)carve((size_t)cD * cD * 2);
    bf16_t* wv16  = (bf16_t*)carve((size_t)cD * cD * 2);
    bf16_t* w116  = (bf16_t*)carve((size_t)cFF * cD * 2);
    bf16_t* w216  = (bf16_t*)carve((size_t)cD * cFF * 2);
    bf16_t* q16   = (bf16_t*)carve((size_t)M * cD * 2);
    bf16_t* k16   = (bf16_t*)carve((size_t)M * cD * 2);
    bf16_t* vT16  = (bf16_t*)carve((size_t)cB * cD * cS * 2); // [b][D][S]
    float*  scF   = (float*) carve((size_t)cB * cS * cS * 4); // reused as h16
    bf16_t* p16   = (bf16_t*)carve((size_t)cB * cS * cS * 2);
    bf16_t* attn16 = emb16;                       // reuse: emb dead after QKV
    bf16_t* h16    = (bf16_t*)scF;                // reuse: scores dead after softmax

    const dim3 blk(256);
    auto cvt = [&](const float* src, bf16_t* dst, size_t n) {
        const int n4 = (int)(n >> 2);
        cvt_f32_bf16<<<dim3((n4 + 255) / 256), blk, 0, stream>>>(src, dst, n4);
    };

    // 1) precision conversion
    cvt(emb, emb16, (size_t)M * cD);
    cvt(Wq,  wq16,  (size_t)cD * cD);
    cvt(Wk,  wk16,  (size_t)cD * cD);
    cvt(Wv,  wv16,  (size_t)cD * cD);
    cvt(W1,  w116,  (size_t)cFF * cD);
    cvt(W2,  w216,  (size_t)cD * cFF);

    // 2) QKV projections (V stored transposed per batch for the P@V GEMM)
    gemm_bf16_wmma<bf16_t, false, false>
        <<<dim3(cD / 128, M / 128, 1), blk, 0, stream>>>(
            emb16, wq16, bq, q16, M, cD, cD, cD, cD, cD, 0, 0, 0, 1.f);
    gemm_bf16_wmma<bf16_t, false, false>
        <<<dim3(cD / 128, M / 128, 1), blk, 0, stream>>>(
            emb16, wk16, bk, k16, M, cD, cD, cD, cD, cD, 0, 0, 0, 1.f);
    gemm_bf16_wmma<bf16_t, false, true>
        <<<dim3(cD / 128, M / 128, 1), blk, 0, stream>>>(
            emb16, wv16, bv, vT16, M, cD, cD, cD, cD, cD, 0, 0, 0, 1.f);

    // 3) scores = Q @ K^T / sqrt(D)   (batched over B)
    const float inv_sqrt_d = 1.0f / sqrtf((float)cD);
    gemm_bf16_wmma<float, false, false>
        <<<dim3(cS / 128, cS / 128, cB), blk, 0, stream>>>(
            q16, k16, nullptr, scF, cS, cS, cD, cD, cD, cS,
            (long long)cS * cD, (long long)cS * cD, (long long)cS * cS,
            inv_sqrt_d);

    // 4) softmax rows -> P (bf16)
    softmax_rows<<<dim3(cB * cS), blk, 0, stream>>>(scF, p16);

    // 5) attn = P @ V   (W = V^T per batch, row-major [D][S])
    gemm_bf16_wmma<bf16_t, false, false>
        <<<dim3(cD / 128, cS / 128, cB), blk, 0, stream>>>(
            p16, vT16, nullptr, attn16, cS, cD, cS, cS, cS, cD,
            (long long)cS * cS, (long long)cD * cS, (long long)cS * cD, 1.f);

    // 6) FFN1: h = relu(attn @ W1^T + b1)
    gemm_bf16_wmma<bf16_t, true, false>
        <<<dim3(cFF / 128, M / 128, 1), blk, 0, stream>>>(
            attn16, w116, b1, h16, M, cFF, cD, cD, cD, cFF, 0, 0, 0, 1.f);

    // 7) FFN2: out = h @ W2^T + b2  (f32 directly into d_out)
    gemm_bf16_wmma<float, false, false>
        <<<dim3(cD / 128, M / 128, 1), blk, 0, stream>>>(
            h16, w216, b2, (float*)d_out, M, cD, cFF, cFF, cFF, cD, 0, 0, 0, 1.f);
}
